// CausalSelfAttention_66606352826775
// MI455X (gfx1250) — matrix operand order repaired and checked by reading, not compiled, not run on hardware
//
#include <hip/hip_runtime.h>
#include <hip/hip_bf16.h>

// ---------------------------------------------------------------------------
// CDNA5 (gfx1250) causal self-attention: bf16 WMMA + Tensor Data Mover (TDM).
//   B=4, T=2048, C=1024, H=16, D=64
// ---------------------------------------------------------------------------

typedef __bf16          v16bf   __attribute__((ext_vector_type(16)));
typedef float           v8f     __attribute__((ext_vector_type(8)));
typedef unsigned short  v16us   __attribute__((ext_vector_type(16)));
typedef unsigned short  v8us    __attribute__((ext_vector_type(8)));
typedef unsigned short  v4us    __attribute__((ext_vector_type(4)));
typedef unsigned int    uint32x4 __attribute__((ext_vector_type(4)));
typedef int             int32x8  __attribute__((ext_vector_type(8)));
typedef int             int32x4  __attribute__((ext_vector_type(4)));

#define T_SEQ 2048
#define C_DIM 1024
#define H_NUM 16
#define D_HEAD 64

__device__ __forceinline__ unsigned short f32bf(float f) {
    unsigned int u = __builtin_bit_cast(unsigned int, f);
    return (unsigned short)((u + 0x7FFFu + ((u >> 16) & 1u)) >> 16);
}

// ---------------------------------------------------------------------------
// TDM: 2D tile load global -> LDS with LDS row padding.
//   tile_w elems/row (bf16), tile_h rows, row_stride elems between rows.
//   pad_int_code: bytes-before-pad = 8 << code ; pad_amt_code: pad = (code+1) DW
// D# layout per CDNA5 ISA.
// ---------------------------------------------------------------------------
__device__ __forceinline__ void tdm_load_2d(unsigned lds_addr, const void* gaddr,
                                            unsigned tile_w, unsigned tile_h,
                                            unsigned row_stride,
                                            int pad_int_code, int pad_amt_code) {
    unsigned long long ga = (unsigned long long)(uintptr_t)gaddr;
    uint32x4 g0;
    g0[0] = 1u;                                               // count=1 (valid)
    g0[1] = lds_addr;                                         // LDS byte addr
    g0[2] = (unsigned)ga;                                     // global_addr lo
    g0[3] = (unsigned)((ga >> 32) & 0x01FFFFFFu) | 0x80000000u; // ga hi | type=2
    int32x8 g1;
    g1[0] = (1 << 16)                                         // data_size = 2B
          | (1 << 20)                                         // pad_enable
          | (pad_int_code << 22) | (pad_amt_code << 25);
    g1[1] = (int)((tile_w & 0xFFFFu) << 16);                  // tensor_dim0 lo16
    g1[2] = (int)(((tile_w >> 16) & 0xFFFFu) | ((tile_h & 0xFFFFu) << 16));
    g1[3] = (int)(((tile_h >> 16) & 0xFFFFu) | ((tile_w & 0xFFFFu) << 16)); // tile_dim0
    g1[4] = (int)(tile_h & 0xFFFFu);                          // tile_dim1 (dim2=0)
    g1[5] = (int)row_stride;                                  // dim0_stride lo32
    g1[6] = 0;
    g1[7] = 0;
    int32x4 z4 = {};
#if __has_include(<hip/amd_detail/amd_gfx1250_TDM.h>)
    int32x8 z8 = {};
    __builtin_amdgcn_tensor_load_to_lds(g0, g1, z4, z4, z8, 0);
#else
    __builtin_amdgcn_tensor_load_to_lds(g0, g1, z4, z4, 0);
#endif
}

__device__ __forceinline__ unsigned lds_off(const void* p) {
    return (unsigned)(uintptr_t)p;      // low 32 bits of generic addr = LDS offset
}

// A-fragment (16x32 bf16, M = lane%16): two contiguous 8-elem chunks per lane.
__device__ __forceinline__ v16bf ld_frag_a(const unsigned short* p, int row_base,
                                           int stride, int col0) {
    const int lane = threadIdx.x & 31;
    const int row  = row_base + (lane & 15);
    const int c    = col0 + ((lane >> 4) << 3);
    const unsigned short* q = p + row * stride + c;
    v8us lo = *(const v8us*)(q);
    v8us hi = *(const v8us*)(q + 16);
    v16us u;
#pragma unroll
    for (int i = 0; i < 8; ++i) { u[i] = lo[i]; u[i + 8] = hi[i]; }
    return __builtin_bit_cast(v16bf, u);
}

// B-fragment (32x16 bf16, N = lane%16): two contiguous 8-elem chunks per lane.
__device__ __forceinline__ v16bf ld_frag_b(const unsigned short* p, int row_base,
                                           int stride, int col0) {
    const int lane = threadIdx.x & 31;
    const int row  = row_base + (lane & 15);
    const int c    = col0 + ((lane >> 4) << 3);
    const unsigned short* q = p + row * stride + c;
    v8us lo = *(const v8us*)(q);
    v8us hi = *(const v8us*)(q + 16);
    v16us u;
#pragma unroll
    for (int i = 0; i < 8; ++i) { u[i] = lo[i]; u[i + 8] = hi[i]; }
    return __builtin_bit_cast(v16bf, u);
}

__device__ __forceinline__ v8f wmma_bf16(v16bf a, v16bf b, v8f c) {
    return __builtin_amdgcn_wmma_f32_16x16x32_bf16(false, a, false, b,
                                                   (short)0, c, false, false);
}

// ---------------------------------------------------------------------------
// Kernel 1: qkv = x @ w_attn^T + b_attn -> Q[B,H,T,D]*scale, K[B,H,T,D],
//           Vt[B,H,D,T]  (bf16). 128 thr, 64x64 tile, K-step 32.
// ---------------------------------------------------------------------------
__global__ __launch_bounds__(128)
void qkv_gemm_kernel(const float* __restrict__ x, const float* __restrict__ w,
                     const float* __restrict__ bias,
                     unsigned short* __restrict__ Q, unsigned short* __restrict__ K,
                     unsigned short* __restrict__ Vt) {
    const int n0   = blockIdx.x * 64;
    const int m0   = blockIdx.y * 64;
    const int tid  = threadIdx.x;
    const int lane = tid & 31;
    const int wave = tid >> 5;
    const int half = lane >> 4;
    const int col  = lane & 15;
    const int STR  = 40;

    __shared__ unsigned short Xs[64 * 40];
    __shared__ unsigned short Ws[64 * 40];

    v8f acc[4] = {};

    for (int kk = 0; kk < C_DIM; kk += 32) {
        float4 fx[4], fw[4];
#pragma unroll
        for (int i = 0; i < 4; ++i) {          // 512 chunks of 4 f32, 4/thread
            const int c = tid + i * 128;
            const int row = c >> 3, cc = (c & 7) * 4;
            fx[i] = *(const float4*)(x + (size_t)(m0 + row) * C_DIM + kk + cc);
            fw[i] = *(const float4*)(w + (size_t)(n0 + row) * C_DIM + kk + cc);
        }
        if (kk + 32 < C_DIM) {                 // prefetch next K-slab into GL2
            const int c = tid, row = c >> 3, cc = (c & 7) * 4;
            __builtin_prefetch(x + (size_t)(m0 + row) * C_DIM + kk + 32 + cc, 0, 3);
            __builtin_prefetch(w + (size_t)(n0 + row) * C_DIM + kk + 32 + cc, 0, 3);
        }
#pragma unroll
        for (int i = 0; i < 4; ++i) {
            const int c = tid + i * 128;
            const int row = c >> 3, cc = (c & 7) * 4;
            v4us hx; hx[0] = f32bf(fx[i].x); hx[1] = f32bf(fx[i].y);
                     hx[2] = f32bf(fx[i].z); hx[3] = f32bf(fx[i].w);
            *(v4us*)(Xs + row * STR + cc) = hx;
            v4us hw; hw[0] = f32bf(fw[i].x); hw[1] = f32bf(fw[i].y);
                     hw[2] = f32bf(fw[i].z); hw[3] = f32bf(fw[i].w);
            *(v4us*)(Ws + row * STR + cc) = hw;
        }
        __syncthreads();
        v16bf a = ld_frag_a(Xs, wave * 16, STR, 0);
#pragma unroll
        for (int nb = 0; nb < 4; ++nb) {
            v16bf b = ld_frag_b(Ws, nb * 16, STR, 0);
            acc[nb] = wmma_bf16(a, b, acc[nb]);
        }
        __syncthreads();
    }

#pragma unroll
    for (int nb = 0; nb < 4; ++nb) {
        const int n  = n0 + nb * 16 + col;
        const float bv = bias[n];
        const int sec = n >> 10;               // 0:Q 1:K 2:V
        const int ch  = n & 1023;
        const int h   = ch >> 6;
        const int d   = ch & 63;
#pragma unroll
        for (int r = 0; r < 8; ++r) {
            const int m  = m0 + wave * 16 + r + 8 * half;
            const int b_ = m >> 11;
            const int t  = m & (T_SEQ - 1);
            const size_t bh = (size_t)(b_ * H_NUM + h);
            const float v = acc[nb][r] + bv;
            if (sec == 0)
                Q[(bh * T_SEQ + t) * D_HEAD + d] = f32bf(v * 0.125f);
            else if (sec == 1)
                K[(bh * T_SEQ + t) * D_HEAD + d] = f32bf(v);
            else
                Vt[(bh * D_HEAD + d) * T_SEQ + t] = f32bf(v);
        }
    }
}

// ---------------------------------------------------------------------------
// Kernel 2: flash attention, TDM-fed, double-buffered K/V tiles.
// Block = (b,h, 64-row q tile), 4 waves; each wave owns 16 q rows.
// ---------------------------------------------------------------------------
__global__ __launch_bounds__(128)
void flash_attn_kernel(const unsigned short* __restrict__ Q,
                       const unsigned short* __restrict__ K,
                       const unsigned short* __restrict__ Vt,
                       unsigned short* __restrict__ Y) {
    const int qt   = blockIdx.x;
    const int bh   = blockIdx.y;
    const int tid  = threadIdx.x;
    const int lane = tid & 31;
    const int wave = tid >> 5;
    const int half = lane >> 4;
    const int col  = lane & 15;
    const int STR  = 72;                       // 64 + 4DW pad from TDM

    __shared__ unsigned short Qs[64 * 72];
    __shared__ unsigned short Ks[2][64 * 72];
    __shared__ unsigned short Vs[2][64 * 72];
    __shared__ unsigned short Ps[4][16 * 72];

    const int ntiles = qt + 1;
    // TDM issue: wave 0 only (EXEC is ignored by TDM; one wave = one transfer)
    if (wave == 0) {
        tdm_load_2d(lds_off(Qs),
                    Q + ((size_t)bh * T_SEQ + (size_t)qt * 64) * D_HEAD,
                    64, 64, D_HEAD, /*128B*/4, /*4DW*/3);
        tdm_load_2d(lds_off(Ks[0]),
                    K + ((size_t)bh * T_SEQ + 0) * D_HEAD,
                    64, 64, D_HEAD, 4, 3);
        tdm_load_2d(lds_off(Vs[0]),
                    Vt + ((size_t)bh * D_HEAD) * T_SEQ + 0,
                    64, 64, T_SEQ, 4, 3);
    }

    float m_r[8], l_r[8];
#pragma unroll
    for (int r = 0; r < 8; ++r) { m_r[r] = -1e30f; l_r[r] = 0.0f; }
    v8f acc[4] = {};
    const int q_row_base = qt * 64 + wave * 16 + 8 * half;

    for (int it = 0; it < ntiles; ++it) {
        const int kt  = it * 64;
        const int cur = it & 1;
        if (wave == 0) {
            if (it + 1 < ntiles) {             // prefetch next tile into alt buf
                const int nk = kt + 64, nxt = cur ^ 1;
                tdm_load_2d(lds_off(Ks[nxt]),
                            K + ((size_t)bh * T_SEQ + nk) * D_HEAD,
                            64, 64, D_HEAD, 4, 3);
                tdm_load_2d(lds_off(Vs[nxt]),
                            Vt + ((size_t)bh * D_HEAD) * T_SEQ + nk,
                            64, 64, T_SEQ, 4, 3);
                __builtin_amdgcn_s_wait_tensorcnt(2);  // current tile done
            } else {
                __builtin_amdgcn_s_wait_tensorcnt(0);
            }
        }
        __syncthreads();                       // publish TDM-written LDS

        // S = Q K^T (16 x 64 per wave)
        v8f s[4] = {};
#pragma unroll
        for (int ks = 0; ks < 2; ++ks) {
            v16bf a = ld_frag_a(Qs, wave * 16, STR, ks * 32);
#pragma unroll
            for (int nb = 0; nb < 4; ++nb) {
                v16bf b = ld_frag_b(Ks[cur], nb * 16, STR, ks * 32);
                s[nb] = wmma_bf16(a, b, s[nb]);
            }
        }

        if (kt + 64 > qt * 64) {               // diagonal tile: causal mask
#pragma unroll
            for (int nb = 0; nb < 4; ++nb) {
                const int kg = kt + nb * 16 + col;
#pragma unroll
                for (int r = 0; r < 8; ++r)
                    if (kg > q_row_base + r) s[nb][r] = -1e30f;
            }
        }

        // online softmax (row = r + 8*half, replicated over the 16-lane half)
#pragma unroll
        for (int r = 0; r < 8; ++r) {
            float mx = fmaxf(fmaxf(s[0][r], s[1][r]), fmaxf(s[2][r], s[3][r]));
            mx = fmaxf(mx, __shfl_xor(mx, 1));
            mx = fmaxf(mx, __shfl_xor(mx, 2));
            mx = fmaxf(mx, __shfl_xor(mx, 4));
            mx = fmaxf(mx, __shfl_xor(mx, 8));
            const float mnew  = fmaxf(m_r[r], mx);
            const float scale = __expf(m_r[r] - mnew);
            float rs = 0.0f;
#pragma unroll
            for (int nb = 0; nb < 4; ++nb) {
                const float p = __expf(s[nb][r] - mnew);
                s[nb][r] = p;
                rs += p;
            }
            rs += __shfl_xor(rs, 1);
            rs += __shfl_xor(rs, 2);
            rs += __shfl_xor(rs, 4);
            rs += __shfl_xor(rs, 8);
            l_r[r] = l_r[r] * scale + rs;
            m_r[r] = mnew;
#pragma unroll
            for (int nb = 0; nb < 4; ++nb) acc[nb][r] *= scale;
        }

        // P -> LDS (bf16 16x64, wave-private)
#pragma unroll
        for (int nb = 0; nb < 4; ++nb)
#pragma unroll
            for (int r = 0; r < 8; ++r)
                Ps[wave][(r + 8 * half) * STR + nb * 16 + col] = f32bf(s[nb][r]);
        __syncthreads();

        // O += P V
#pragma unroll
        for (int ks = 0; ks < 2; ++ks) {
            v16bf a = ld_frag_a(Ps[wave], 0, STR, ks * 32);
#pragma unroll
            for (int nb = 0; nb < 4; ++nb) {
                v16bf b = ld_frag_b(Vs[cur], nb * 16, STR, ks * 32);
                acc[nb] = wmma_bf16(a, b, acc[nb]);
            }
        }
        __syncthreads();                       // done reading this buffer
    }

    // epilogue: normalize, write y (bf16, [B,T,C])
    const int b_ = bh >> 4;
    const int h  = bh & 15;
#pragma unroll
    for (int r = 0; r < 8; ++r) {
        const float inv = 1.0f / l_r[r];
        const int t = qt * 64 + wave * 16 + r + 8 * half;
#pragma unroll
        for (int nb = 0; nb < 4; ++nb) {
            const int d = nb * 16 + col;
            Y[((size_t)(b_ * T_SEQ + t)) * C_DIM + h * D_HEAD + d] =
                f32bf(acc[nb][r] * inv);
        }
    }
}

// ---------------------------------------------------------------------------
// Kernel 3: out = y @ w_proj^T + b_proj   (y bf16 via TDM, out f32)
// ---------------------------------------------------------------------------
__global__ __launch_bounds__(128)
void proj_gemm_kernel(const unsigned short* __restrict__ Ybf,
                      const float* __restrict__ w, const float* __restrict__ bias,
                      float* __restrict__ out) {
    const int n0   = blockIdx.x * 64;
    const int m0   = blockIdx.y * 64;
    const int tid  = threadIdx.x;
    const int lane = tid & 31;
    const int wave = tid >> 5;
    const int half = lane >> 4;
    const int col  = lane & 15;
    const int STR  = 40;                       // 32 + 4DW pad from TDM

    __shared__ unsigned short Xs[64 * 40];
    __shared__ unsigned short Ws[64 * 40];

    v8f acc[4] = {};

    for (int kk = 0; kk < C_DIM; kk += 32) {
        // TDM: bf16 A tile (64 rows x 32), pad 16B every 64B -> stride 40
        if (wave == 0)
            tdm_load_2d(lds_off(Xs), Ybf + (size_t)m0 * C_DIM + kk,
                        32, 64, C_DIM, /*64B*/3, /*4DW*/3);
        // manual: f32 W -> bf16 (TDM cannot convert)
        float4 fw[4];
#pragma unroll
        for (int i = 0; i < 4; ++i) {
            const int c = tid + i * 128;
            const int row = c >> 3, cc = (c & 7) * 4;
            fw[i] = *(const float4*)(w + (size_t)(n0 + row) * C_DIM + kk + cc);
        }
#pragma unroll
        for (int i = 0; i < 4; ++i) {
            const int c = tid + i * 128;
            const int row = c >> 3, cc = (c & 7) * 4;
            v4us hw; hw[0] = f32bf(fw[i].x); hw[1] = f32bf(fw[i].y);
                     hw[2] = f32bf(fw[i].z); hw[3] = f32bf(fw[i].w);
            *(v4us*)(Ws + row * STR + cc) = hw;
        }
        if (wave == 0) __builtin_amdgcn_s_wait_tensorcnt(0);
        __syncthreads();
        v16bf a = ld_frag_a(Xs, wave * 16, STR, 0);
#pragma unroll
        for (int nb = 0; nb < 4; ++nb) {
            v16bf b = ld_frag_b(Ws, nb * 16, STR, 0);
            acc[nb] = wmma_bf16(a, b, acc[nb]);
        }
        __syncthreads();
    }

#pragma unroll
    for (int nb = 0; nb < 4; ++nb) {
        const int n = n0 + nb * 16 + col;
        const float bv = bias[n];
#pragma unroll
        for (int r = 0; r < 8; ++r) {
            const int m = m0 + wave * 16 + r + 8 * half;
            out[(size_t)m * C_DIM + n] = acc[nb][r] + bv;
        }
    }
}

// ---------------------------------------------------------------------------
extern "C" void kernel_launch(void* const* d_in, const int* in_sizes, int n_in,
                              void* d_out, int out_size, void* d_ws, size_t ws_size,
                              hipStream_t stream) {
    (void)in_sizes; (void)n_in; (void)out_size; (void)ws_size;

    const float* x      = (const float*)d_in[0];
    const float* w_attn = (const float*)d_in[1];
    const float* b_attn = (const float*)d_in[2];
    const float* w_proj = (const float*)d_in[3];
    const float* b_proj = (const float*)d_in[4];
    float* out = (float*)d_out;

    unsigned short* ws = (unsigned short*)d_ws;
    const size_t SEG = (size_t)4 * T_SEQ * C_DIM;   // 8,388,608 elements
    unsigned short* Q  = ws;
    unsigned short* K  = ws + SEG;
    unsigned short* Vt = ws + 2 * SEG;
    unsigned short* Y  = ws + 3 * SEG;

    qkv_gemm_kernel<<<dim3(3 * C_DIM / 64, 4 * T_SEQ / 64), 128, 0, stream>>>(
        x, w_attn, b_attn, Q, K, Vt);

    flash_attn_kernel<<<dim3(T_SEQ / 64, 4 * H_NUM), 128, 0, stream>>>(Q, K, Vt, Y);

    proj_gemm_kernel<<<dim3(C_DIM / 64, 4 * T_SEQ / 64), 128, 0, stream>>>(
        Y, w_proj, b_proj, out);
}
